// RIME_73847667687923
// MI455X (gfx1250) — matrix operand order repaired and checked by reading, not compiled, not run on hardware
//
#include <hip/hip_runtime.h>

// RIME visibilities on gfx1250 via dense WMMA.
//
// vis[p,q,k,t,f] = sum_s sky[pq,f,s] * beam[p,m1(k),t,f,s] * beam[q,m2(k),t,f,s]
//                        * exp(-2i*pi*freq_f*tau[k,t,s])
//
// Only Nbeam=2 beam models exist, so the weight
//   W[pq,k,s] = sky[pq,s]*beam[p,m1(k),s]*beam[q,m2(k),s]
// takes only 4 values per pq, indexed by combo(k) = 2*m1+m2. We build a
// 16-row matrix T[r=(pq,combo)][s] shared by ALL baselines and compute
//   D[16,16] += T[16, Ks] x F[Ks, 16]        (v_wmma_f32_16x16x32_f16)
// where the 16 B columns are (8 baselines) x (cos, sin). Each baseline
// extracts its 4 (pq) outputs from the rows matching its combo at writeout.
//
// Fringe: HW cos works in revolutions, so phase/(2pi) = -(freq/c)*(blvec.svec)
// and sin is folded in as a -0.25-revolution shift on odd columns — a single
// v_cos_f32 per fringe element, no sin/cos pair + cndmask.

typedef __attribute__((ext_vector_type(16))) _Float16 v16h;
typedef __attribute__((ext_vector_type(8)))  _Float16 v8h;
typedef __attribute__((ext_vector_type(8)))  float    v8f;

#define NPOL   2
#define NBEAM  2
#define NANTS  12
#define NTIMES 8
#define NFREQS 64
#define NSRC   512
#define TPAD   520            // 512 + 8 halfs: row stride 1040B -> rotates LDS banks
#define NTHREADS 288          // 9 waves * 32 : ceil(66/8) baseline tiles

__global__ __launch_bounds__(NTHREADS) void rime_wmma_kernel(
    const float* __restrict__ sky,       // [2,2,64,512]
    const float* __restrict__ beam,      // [2,2,8,64,512]
    const float* __restrict__ antpos,    // [12,3]
    const float* __restrict__ svec,      // [8,3,512]
    const float* __restrict__ freqs,     // [64]
    const int*   __restrict__ ant2model, // [12]
    const int*   __restrict__ bls,       // [66,2]
    float* __restrict__ out,             // [2,2,66,8,64] complex64 (re,im interleaved)
    int nbl)
{
    __shared__ __align__(16) _Float16 T[16][TPAD]; // rows r = 8p+4q+2m1+m2
    __shared__ __align__(16) float    sv[3][NSRC];

    const int f   = blockIdx.x;
    const int t   = blockIdx.y;
    const int tid = threadIdx.x;

    // ---- Stage T (beam-combo * sky weights) and this time's source vectors ----
    for (int idx = tid; idx < 16 * NSRC; idx += NTHREADS) {
        const int r = idx >> 9, s = idx & (NSRC - 1);
        const int p = r >> 3, q = (r >> 2) & 1, m1 = (r >> 1) & 1, m2 = r & 1;
        const float sk = sky [((p * 2 + q ) * NFREQS + f) * NSRC + s];
        const float b1 = beam[(((p * 2 + m1) * NTIMES + t) * NFREQS + f) * NSRC + s];
        const float b2 = beam[(((q * 2 + m2) * NTIMES + t) * NFREQS + f) * NSRC + s];
        T[r][s] = (_Float16)(sk * b1 * b2);
    }
    for (int idx = tid; idx < 3 * NSRC; idx += NTHREADS) {
        const int c = idx >> 9, s = idx & (NSRC - 1);
        sv[c][s] = svec[(t * 3 + c) * NSRC + s];
    }
    __syncthreads();

    // ---- Per-lane roles (wave = 8-baseline column tile) ----
    const int wave = tid >> 5;
    const int lane = tid & 31;
    const int h    = lane >> 4;     // lane half (selects K/M sub-blocks)
    const int n    = lane & 15;     // B/D column
    const int lb   = n >> 1;        // local baseline in tile
    const int comp = n & 1;         // 0 = cos(re), 1 = sin(im)
    int k = wave * 8 + lb;
    const bool valid = (k < nbl);
    k = valid ? k : (nbl - 1);      // clamp; WMMA needs full EXEC, discard later

    const int   a1 = bls[2 * k + 0], a2 = bls[2 * k + 1];
    const float bx = antpos[a2 * 3 + 0] - antpos[a1 * 3 + 0];
    const float by = antpos[a2 * 3 + 1] - antpos[a1 * 3 + 1];
    const float bz = antpos[a2 * 3 + 2] - antpos[a1 * 3 + 2];
    // phase in revolutions: u = -(freq/c) * (blvec . svec); fold coef into blvec
    const float coef  = -freqs[f] * (1.0f / 299792458.0f);
    const float cbx = coef * bx, cby = coef * by, cbz = coef * bz;
    // sin(2*pi*u) = cos(2*pi*(u - 1/4)): odd columns get a -0.25 rev shift,
    // seeded as the FMA-chain accumulator start -> one v_cos_f32 per element.
    const float mshift = comp ? -0.25f : 0.0f;
    const int   combo  = ant2model[a1] * 2 + ant2model[a2];

    const _Float16* Trow = &T[lane & 15][0];   // A row M = lane%16

    v8f acc = {};
    #pragma unroll
    for (int ck = 0; ck < NSRC / 32; ++ck) {
        // A fragment (16x32 f16): K = 16h + e
        const int s0 = ck * 32 + 16 * h;
        v16h a;
        {
            const v8h lo = *(const v8h*)(Trow + s0);
            const v8h hi = *(const v8h*)(Trow + s0 + 8);
            #pragma unroll
            for (int e = 0; e < 8; ++e) { a[e] = lo[e]; a[8 + e] = hi[e]; }
        }
        // B fragment (32x16 f16): K = 16h + e, column n -> fringe(comp) of baseline lb
        const int sb = ck * 32 + 16 * h;
        v16h b;
        #pragma unroll
        for (int e = 0; e < 16; ++e) {
            const int s = sb + e;
            const float u = __builtin_fmaf(cbx, sv[0][s],
                            __builtin_fmaf(cby, sv[1][s],
                            __builtin_fmaf(cbz, sv[2][s], mshift)));
            b[e] = (_Float16)__builtin_amdgcn_cosf(u);   // cos(2*pi*u)
        }
        acc = __builtin_amdgcn_wmma_f32_16x16x32_f16(
                  false, a, false, b, (short)0, acc, false, false);
    }

    if (valid) {
        // D layout: VGPR r holds M = r + 8h for this lane's column n.
        // Column n needs rows M = 4*pq + combo; for pq in {2h, 2h+1} the
        // VGPR indices collapse to combo and 4+combo for BOTH halves.
        const float r0 = (combo & 2) ? ((combo & 1) ? acc[3] : acc[2])
                                     : ((combo & 1) ? acc[1] : acc[0]);
        const float r1 = (combo & 2) ? ((combo & 1) ? acc[7] : acc[6])
                                     : ((combo & 1) ? acc[5] : acc[4]);
        const int  pq0   = 2 * h;            // p*2+q
        const long base0 = ((((long)pq0       * nbl + k) * NTIMES + t) * NFREQS + f) * 2 + comp;
        const long base1 = ((((long)(pq0 + 1) * nbl + k) * NTIMES + t) * NFREQS + f) * 2 + comp;
        out[base0] = r0;
        out[base1] = r1;
    }
}

extern "C" void kernel_launch(void* const* d_in, const int* in_sizes, int n_in,
                              void* d_out, int out_size, void* d_ws, size_t ws_size,
                              hipStream_t stream) {
    const float* sky       = (const float*)d_in[0];
    const float* beam      = (const float*)d_in[1];
    const float* antpos    = (const float*)d_in[2];
    const float* svec      = (const float*)d_in[3];
    const float* freqs     = (const float*)d_in[4];
    const int*   ant2model = (const int*)d_in[5];
    const int*   bls       = (const int*)d_in[6];
    const int    nbl       = in_sizes[6] / 2;   // 66

    dim3 grid(NFREQS, NTIMES, 1);               // one workgroup per (t, f)
    dim3 block(NTHREADS, 1, 1);
    rime_wmma_kernel<<<grid, block, 0, stream>>>(
        sky, beam, antpos, svec, freqs, ant2model, bls, (float*)d_out, nbl);
}
